// GSA_22445499089534
// MI455X (gfx1250) — compile-verified, matrix operand order checked by reading, not run on hardware
//
#include <hip/hip_runtime.h>
#include <math.h>

typedef __attribute__((ext_vector_type(16))) _Float16 v16h;
typedef __attribute__((ext_vector_type(8)))  float    v8f;

#define BB 16
#define HH 128
#define WW 128
#define CC 192
#define LL (HH * WW)        // 16384 sequence length per batch
#define NN 16               // state size
#define CHUNK 64
#define NCH (LL / CHUNK)    // 256 chunks per batch

// ---------------------------------------------------------------------------
// Kernel 1: channel mean-pool via WMMA.
// One wave handles 16 pixels. A-tile = 16 pixels x 32 channels (fp16),
// B = all-ones fp16 (uniform over K, so K-permutation in the A layout is
// irrelevant), f32 accumulator across 6 K-chunks (192 = 6*32).
// D layout (32-bit C/D 16x16): lane 0 vgpr r = D[r][0], lane 16 vgpr r = D[8+r][0].
// ---------------------------------------------------------------------------
__global__ __launch_bounds__(128) void pool_wmma(const float* __restrict__ x,
                                                 float* __restrict__ u) {
    const int lane  = threadIdx.x & 31;
    const int wave  = threadIdx.x >> 5;
    const long p0   = (long)blockIdx.x * 64 + (long)wave * 16;  // 16 pixels / wave
    const int row   = lane & 15;
    const int khalf = (lane >> 4) * 16;    // lanes 0-15: K 0..15, lanes 16-31: K 16..31

    v16h bm;
#pragma unroll
    for (int i = 0; i < 16; ++i) bm[i] = (_Float16)1.0f;

    v8f acc = {};
#pragma unroll
    for (int kk = 0; kk < 6; ++kk) {
        const float4* px4 = (const float4*)(x + (p0 + row) * CC + kk * 32 + khalf);
        v16h am;
#pragma unroll
        for (int j = 0; j < 4; ++j) {
            float4 v = px4[j];
            am[4 * j + 0] = (_Float16)v.x;
            am[4 * j + 1] = (_Float16)v.y;
            am[4 * j + 2] = (_Float16)v.z;
            am[4 * j + 3] = (_Float16)v.w;
        }
        // (neg_a, A, neg_b, B, c_mod, C, reuse_a, reuse_b)
        acc = __builtin_amdgcn_wmma_f32_16x16x32_f16(false, am, false, bm,
                                                     (short)0, acc, false, false);
    }

    const float inv = 1.0f / (float)CC;
    if (lane == 0) {
#pragma unroll
        for (int r = 0; r < 8; ++r) u[p0 + r] = acc[r] * inv;
    } else if (lane == 16) {
#pragma unroll
        for (int r = 0; r < 8; ++r) u[p0 + 8 + r] = acc[r] * inv;
    }
}

// ---------------------------------------------------------------------------
// Kernel 2: chunked parallel selective scan (one block per batch element).
// Phase 1: each thread scans a 64-step chunk with h0=0, recording
//          cumP[n] = prod(dA) and local state s[n].
// Phase 2: 16 threads (one per state) serially combine 256 chunk summaries.
// Phase 3: each thread re-scans its chunk from the correct incoming state and
//          emits gate = sigmoid(sum_n h[n]*C_t[n]).
// ---------------------------------------------------------------------------
__global__ __launch_bounds__(256) void s6_scan(const float* __restrict__ U,
                                               const float* __restrict__ A,
                                               const float* __restrict__ Wd,
                                               const float* __restrict__ bd,
                                               const float* __restrict__ WB,
                                               const float* __restrict__ bBp,
                                               const float* __restrict__ WC,
                                               const float* __restrict__ bCp,
                                               float* __restrict__ gate) {
    __shared__ float sP[NCH][NN + 1];   // +1 pad: avoid bank conflicts
    __shared__ float sS[NCH][NN + 1];
    __shared__ float sH[NCH][NN + 1];

    const int tid   = threadIdx.x;
    const int b     = blockIdx.x;
    const long base = (long)b * LL + (long)tid * CHUNK;

    const float wd  = Wd[0];
    const float bdv = bd[0];
    float Ar[NN], wB[NN], bB[NN], wC[NN], bC[NN];
#pragma unroll
    for (int n = 0; n < NN; ++n) {
        Ar[n] = A[n]; wB[n] = WB[n]; bB[n] = bBp[n]; wC[n] = WC[n]; bC[n] = bCp[n];
    }

    // Phase 1 --------------------------------------------------------------
    float cp[NN], s[NN];
#pragma unroll
    for (int n = 0; n < NN; ++n) { cp[n] = 1.0f; s[n] = 0.0f; }
    for (int t = 0; t < CHUNK; ++t) {
        float uv    = U[base + t];
        float z     = wd * uv + bdv;
        float delta = (z > 20.0f) ? z : log1pf(expf(z));   // softplus
        float du    = delta * uv;
#pragma unroll
        for (int n = 0; n < NN; ++n) {
            float dA  = expf(delta * Ar[n]);
            float dBu = du * (wB[n] * uv + bB[n]);
            s[n]  = dA * s[n] + dBu;
            cp[n] *= dA;
        }
    }
#pragma unroll
    for (int n = 0; n < NN; ++n) { sP[tid][n] = cp[n]; sS[tid][n] = s[n]; }
    __syncthreads();

    // Phase 2 --------------------------------------------------------------
    if (tid < NN) {
        float h = 0.0f;
        for (int c = 0; c < NCH; ++c) {
            sH[c][tid] = h;
            h = sP[c][tid] * h + sS[c][tid];
        }
    }
    __syncthreads();

    // Phase 3 --------------------------------------------------------------
    float h[NN];
#pragma unroll
    for (int n = 0; n < NN; ++n) h[n] = sH[tid][n];
    for (int t = 0; t < CHUNK; ++t) {
        float uv    = U[base + t];
        float z     = wd * uv + bdv;
        float delta = (z > 20.0f) ? z : log1pf(expf(z));
        float du    = delta * uv;
        float y     = 0.0f;
#pragma unroll
        for (int n = 0; n < NN; ++n) {
            float dA  = expf(delta * Ar[n]);
            float dBu = du * (wB[n] * uv + bB[n]);
            h[n] = dA * h[n] + dBu;
            y   += h[n] * (wC[n] * uv + bC[n]);
        }
        gate[base + t] = 1.0f / (1.0f + expf(-y));
    }
}

// ---------------------------------------------------------------------------
// Kernel 3: out = x * gate (gate broadcast over 192 channels), float4 streams.
// ---------------------------------------------------------------------------
__global__ __launch_bounds__(256) void gate_mul(const float* __restrict__ x,
                                                const float* __restrict__ gate,
                                                float* __restrict__ out,
                                                long n4) {
    long i = (long)blockIdx.x * blockDim.x + threadIdx.x;
    if (i >= n4) return;
    long p  = i / (CC / 4);            // pixel index (48 float4s per pixel)
    float g = gate[p];
    float4 v = ((const float4*)x)[i];
    v.x *= g; v.y *= g; v.z *= g; v.w *= g;
    ((float4*)out)[i] = v;
}

extern "C" void kernel_launch(void* const* d_in, const int* in_sizes, int n_in,
                              void* d_out, int out_size, void* d_ws, size_t ws_size,
                              hipStream_t stream) {
    const float* x  = (const float*)d_in[0];
    const float* A  = (const float*)d_in[1];
    const float* Wd = (const float*)d_in[2];
    const float* bd = (const float*)d_in[3];
    const float* WB = (const float*)d_in[4];
    const float* bB = (const float*)d_in[5];
    const float* WC = (const float*)d_in[6];
    const float* bC = (const float*)d_in[7];
    float* out = (float*)d_out;

    float* u    = (float*)d_ws;                 // B*L floats = 1 MB
    float* gate = u + (long)BB * LL;            // B*L floats = 1 MB

    // Pass 1: mean-pool via WMMA (16 pixels/wave, 4 waves/block)
    const int pool_blocks = (BB * LL) / 64;     // 4096
    pool_wmma<<<pool_blocks, 128, 0, stream>>>(x, u);

    // Pass 2: chunked parallel S6 scan, one block per batch element
    s6_scan<<<BB, 256, 0, stream>>>(u, A, Wd, bd, WB, bB, WC, bC, gate);

    // Pass 3: gated elementwise multiply
    const long n4 = (long)BB * LL * CC / 4;     // 12,582,912
    gate_mul<<<(int)((n4 + 255) / 256), 256, 0, stream>>>(x, gate, out, n4);
}
